// RWKV_TimeMixing_84413287235986
// MI455X (gfx1250) — matrix operand order repaired and checked by reading, not compiled, not run on hardware
//
#include <hip/hip_runtime.h>
#include <hip/hip_bf16.h>
#include <math.h>

// ---------------- problem constants ----------------
constexpr int Bn = 4, Tn = 4096, Dn = 1024, Hn = 16, Kn = 64;
constexpr size_t XE = (size_t)Bn * Tn * Dn;      // 16,777,216 elements
constexpr int Mrows = Bn * Tn;                   // 16384 GEMM rows
constexpr int Sn = 8, Ln = Tn / Sn;              // scan segments: 8 x 512

// ---------------- WMMA vector types ----------------
typedef _Float16 f16;
typedef f16   v8h  __attribute__((ext_vector_type(8)));
typedef f16   v16h __attribute__((ext_vector_type(16)));
typedef float v8f  __attribute__((ext_vector_type(8)));

// ---------------- workspace layout (bytes, 256-aligned) ----------------
constexpr size_t align256(size_t x) { return (x + 255) & ~(size_t)255; }
constexpr size_t OFF_WR = 0;
constexpr size_t OFF_WK = OFF_WR + align256((size_t)2 * Dn * Dn);
constexpr size_t OFF_WV = OFF_WK + align256((size_t)2 * Dn * Dn);
constexpr size_t OFF_WO = OFF_WV + align256((size_t)2 * Dn * Dn);
constexpr size_t OFF_XR = OFF_WO + align256((size_t)2 * Dn * Dn);
constexpr size_t OFF_XK = OFF_XR + align256(2 * XE);
constexpr size_t OFF_XV = OFF_XK + align256(2 * XE);
constexpr size_t OFF_R  = OFF_XV + align256(2 * XE);
constexpr size_t OFF_K  = OFF_R  + align256(4 * XE);
constexpr size_t OFF_V  = OFF_K  + align256(4 * XE);
constexpr size_t OFF_A  = OFF_V  + align256(4 * XE);
constexpr size_t OFF_Y  = OFF_A  + align256(4 * XE);
// scan scratch
constexpr size_t ZE_BYTES = (size_t)Bn * Hn * Kn * Sn * 2 * 4;   // 256 KB
constexpr size_t OFF_ZE = OFF_Y  + align256(2 * XE);
constexpr size_t OFF_CA = OFF_ZE + align256(ZE_BYTES);
constexpr size_t OFF_PS = OFF_CA + align256(ZE_BYTES);
constexpr size_t OFF_ST = OFF_PS + align256((size_t)Bn * Hn * Sn * 2 * 4);

// ---------------- kernel 1: convert 4 weight matrices f32 -> f16 ----------------
__global__ __launch_bounds__(256) void convw_kernel(
    const float* __restrict__ wr, const float* __restrict__ wk,
    const float* __restrict__ wv, const float* __restrict__ wo,
    f16* __restrict__ wrh, f16* __restrict__ wkh,
    f16* __restrict__ wvh, f16* __restrict__ woh)
{
    size_t i = (size_t)blockIdx.x * blockDim.x + threadIdx.x;
    wrh[i] = (f16)wr[i];
    wkh[i] = (f16)wk[i];
    wvh[i] = (f16)wv[i];
    woh[i] = (f16)wo[i];
}

// ---------------- kernel 2: token-shift mix -> f16 ----------------
__global__ __launch_bounds__(256) void mix_kernel(
    const float* __restrict__ x,
    const float* __restrict__ mr, const float* __restrict__ mk,
    const float* __restrict__ mv,
    f16* __restrict__ xr, f16* __restrict__ xk, f16* __restrict__ xv)
{
    size_t idx = (size_t)blockIdx.x * blockDim.x + threadIdx.x; // over B*T*D
    int d = (int)(idx & (Dn - 1));
    int t = (int)((idx >> 10) & (Tn - 1)); // D = 2^10
    float xc = x[idx];
    float xp = (t == 0) ? 0.0f : x[idx - Dn];
    float r = mr[d], k = mk[d], v = mv[d];
    xr[idx] = (f16)(xc * r + xp * (1.0f - r));
    xk[idx] = (f16)(xc * k + xp * (1.0f - k));
    xv[idx] = (f16)(xc * v + xp * (1.0f - v));
}

// ---------------- kernel 3: WMMA GEMM  C[M,N] = act( A[M,Kd] * W[N,Kd]^T ) ----------------
// Wave computes a 64(M) x 64(N) macro-tile via 4x4 WMMA tiles (intensity ~32 flop/B).
// Block = 256 threads = 8 waves arranged 2(M) x 4(N): block tile = 128M x 256N.
__global__ __launch_bounds__(256) void gemm_f16_wT_kernel(
    const f16* __restrict__ A, const f16* __restrict__ W,
    float* __restrict__ C, int M, int N, int Kd, int act)
{
    const int lane = threadIdx.x & 31;
    const int wave = threadIdx.x >> 5;
    const int wm = wave >> 2;          // 0..1
    const int wn = wave & 3;           // 0..3
    const int la = lane & 15;
    const int lh = lane >> 4;          // 0 or 1

    const int m0 = blockIdx.y * 128 + wm * 64;
    const int n0 = blockIdx.x * 256 + wn * 64;

    v8f acc[4][4];
#pragma unroll
    for (int mi = 0; mi < 4; ++mi)
#pragma unroll
        for (int ni = 0; ni < 4; ++ni)
            acc[mi][ni] = (v8f)(0.0f);

    // A tile mi: lane holds row (m0+mi*16+la), 16 contiguous K at lh*16
    const f16* aptr[4];
#pragma unroll
    for (int mi = 0; mi < 4; ++mi)
        aptr[mi] = A + (size_t)(m0 + mi * 16 + la) * Kd + (lh << 4);
    // B tile ni: lane holds W row (output col) n0+ni*16+la, 16 contiguous K at lh*16
    const f16* bptr = W + (size_t)(n0 + la) * Kd + (lh << 4);

    for (int kk = 0; kk < Kd; kk += 32) {
        // speculative prefetch of A, 256 elements (512B) ahead
#pragma unroll
        for (int mi = 0; mi < 4; ++mi)
            __builtin_prefetch(aptr[mi] + kk + 256, 0, 3);

        union { v16h v; v8h h[2]; } ua[4];
#pragma unroll
        for (int mi = 0; mi < 4; ++mi) {
            ua[mi].h[0] = *(const v8h*)(aptr[mi] + kk);
            ua[mi].h[1] = *(const v8h*)(aptr[mi] + kk + 8);
        }
        v16h bv[4];
#pragma unroll
        for (int ni = 0; ni < 4; ++ni)
            bv[ni] = *(const v16h*)(bptr + (size_t)(ni * 16) * Kd + kk);

#pragma unroll
        for (int mi = 0; mi < 4; ++mi)
#pragma unroll
            for (int ni = 0; ni < 4; ++ni)
                acc[mi][ni] = __builtin_amdgcn_wmma_f32_16x16x32_f16(
                    false, ua[mi].v, false, bv[ni],
                    (short)0, acc[mi][ni], false, false);
    }

    // epilogue: VGPR i -> (M = m0 + mi*16 + i + 8*lh, N = n0 + ni*16 + la)
#pragma unroll
    for (int mi = 0; mi < 4; ++mi) {
        const int mbase = m0 + mi * 16 + (lh << 3);
#pragma unroll
        for (int ni = 0; ni < 4; ++ni) {
            const int n = n0 + ni * 16 + la;
#pragma unroll
            for (int i = 0; i < 8; ++i) {
                float v = acc[mi][ni][i];
                if (act) v = 1.0f / (1.0f + __expf(-v)); // sigmoid for r
                C[(size_t)(mbase + i) * N + n] = v;
            }
        }
    }
}

// ---------------- WKV scan, split-T linear-recurrence decomposition ----------------
// state s_t = w * s_{t-1} + u_t, u_t = (exp(k)*v, exp(k));  per-channel scalar w.
// Phase 1: per-segment local scan from zero -> segment-end partial state.
__global__ __launch_bounds__(64) void scan_phase1(
    const float* __restrict__ kbuf, const float* __restrict__ vbuf,
    const float* __restrict__ time_decay, float* __restrict__ zend)
{
    const int blk = blockIdx.x;          // 0 .. B*H*S-1
    const int seg = blk % Sn;
    const int bh  = blk / Sn;
    const int b = bh / Hn, h = bh % Hn;
    const int kq = threadIdx.x;

    const float w = __expf(-__expf(time_decay[h * Kn + kq]));
    float num = 0.0f, den = 0.0f;
    size_t base = (size_t)b * Tn * Dn + (size_t)seg * Ln * Dn + h * Kn + kq;
    for (int t = 0; t < Ln; ++t) {
        size_t off = base + (size_t)t * Dn;
        float ek = __expf(kbuf[off]);
        num = num * w + ek * vbuf[off];
        den = den * w + ek;
    }
    size_t zi = (((size_t)bh * Kn + kq) * Sn + seg) * 2;
    zend[zi] = num; zend[zi + 1] = den;
}

// Phase 2: serial combine across segments (closed-form w^L), one thread/channel.
__global__ __launch_bounds__(256) void scan_phase2(
    const float* __restrict__ time_decay,
    const float* __restrict__ zend, float* __restrict__ carry)
{
    const int c = blockIdx.x * 256 + threadIdx.x;   // 0 .. B*H*K-1  (= bh*K + kq)
    const int kq = c % Kn;
    const int bh = c / Kn;
    const int h = bh % Hn;
    const float ew = __expf(time_decay[h * Kn + kq]);
    const float wL = __expf(-ew * (float)Ln);       // w^L
    float cn = 0.0f, cd = 0.0f;
    for (int j = 0; j < Sn; ++j) {
        size_t zi = ((size_t)c * Sn + j) * 2;
        carry[zi] = cn; carry[zi + 1] = cd;
        cn = wL * cn + zend[zi];
        cd = wL * cd + zend[zi + 1];
    }
}

// Phase 3: per-segment re-scan with correct entering carry; emits a = r*wkv and
// per-(b,h,segment) GroupNorm partial sums.
__global__ __launch_bounds__(64) void scan_phase3(
    const float* __restrict__ rbuf, const float* __restrict__ kbuf,
    const float* __restrict__ vbuf,
    const float* __restrict__ time_decay, const float* __restrict__ time_first,
    const float* __restrict__ carry,
    float* __restrict__ abuf, float* __restrict__ psum)
{
    const int blk = blockIdx.x;          // 0 .. B*H*S-1
    const int seg = blk % Sn;
    const int bh  = blk / Sn;
    const int b = bh / Hn, h = bh % Hn;
    const int kq = threadIdx.x;
    const int c = bh * Kn + kq;

    const float w  = __expf(-__expf(time_decay[h * Kn + kq]));
    const float eu = __expf(time_first[h * Kn + kq]);

    size_t ci = ((size_t)c * Sn + seg) * 2;
    float num = carry[ci], den = carry[ci + 1];
    float s = 0.0f, ss = 0.0f;

    size_t base = (size_t)b * Tn * Dn + (size_t)seg * Ln * Dn + h * Kn + kq;
    for (int t = 0; t < Ln; ++t) {
        size_t off = base + (size_t)t * Dn;
        float kt = kbuf[off];
        float vt = vbuf[off];
        float rt = rbuf[off];
        float ek = __expf(kt);
        float wkv = (num + eu * ek * vt) / (den + eu * ek + 1e-9f);
        num = num * w + ek * vt;
        den = den * w + ek;
        float a = rt * wkv;
        abuf[off] = a;
        s += a;
        ss += a * a;
    }

    __shared__ float sh1[64];
    __shared__ float sh2[64];
    sh1[kq] = s; sh2[kq] = ss;
    __syncthreads();
    for (int offr = 32; offr > 0; offr >>= 1) {
        if (kq < offr) { sh1[kq] += sh1[kq + offr]; sh2[kq] += sh2[kq + offr]; }
        __syncthreads();
    }
    if (kq == 0) {
        psum[2 * (bh * Sn + seg) + 0] = sh1[0];
        psum[2 * (bh * Sn + seg) + 1] = sh2[0];
    }
}

// Phase 4: reduce per-segment partials -> (mean, rstd) per (b,h). One block.
__global__ __launch_bounds__(64) void scan_stats(
    const float* __restrict__ psum, float* __restrict__ stats)
{
    const int bh = threadIdx.x;          // 0..63
    float s = 0.0f, ss = 0.0f;
    for (int j = 0; j < Sn; ++j) {
        s  += psum[2 * (bh * Sn + j) + 0];
        ss += psum[2 * (bh * Sn + j) + 1];
    }
    const float inv = 1.0f / ((float)Tn * (float)Kn);
    float mean = s * inv;
    float var  = fmaxf(ss * inv - mean * mean, 0.0f);
    stats[2 * bh + 0] = mean;
    stats[2 * bh + 1] = rsqrtf(var + 1e-5f);
}

// ---------------- GroupNorm apply -> f16 ----------------
__global__ __launch_bounds__(256) void norm_kernel(
    const float* __restrict__ abuf, const float* __restrict__ stats,
    const float* __restrict__ gnw, const float* __restrict__ gnb,
    f16* __restrict__ y)
{
    size_t idx = (size_t)blockIdx.x * blockDim.x + threadIdx.x; // over B*T*D
    int d = (int)(idx & (Dn - 1));
    int b = (int)(idx >> 22);                 // T*D = 2^22
    int h = d >> 6;                           // K = 64
    float mean = stats[2 * (b * Hn + h) + 0];
    float rstd = stats[2 * (b * Hn + h) + 1];
    float v = (abuf[idx] - mean) * rstd * gnw[d] + gnb[d];
    y[idx] = (f16)v;
}

// ---------------- launcher ----------------
extern "C" void kernel_launch(void* const* d_in, const int* in_sizes, int n_in,
                              void* d_out, int out_size, void* d_ws, size_t ws_size,
                              hipStream_t stream) {
    const float* x   = (const float*)d_in[0];
    const float* mr  = (const float*)d_in[1];
    const float* mk  = (const float*)d_in[2];
    const float* mv  = (const float*)d_in[3];
    const float* td  = (const float*)d_in[4];
    const float* tf  = (const float*)d_in[5];
    const float* Wr  = (const float*)d_in[6];
    const float* Wk  = (const float*)d_in[7];
    const float* Wv  = (const float*)d_in[8];
    const float* Wo  = (const float*)d_in[9];
    const float* gnw = (const float*)d_in[10];
    const float* gnb = (const float*)d_in[11];
    float* out = (float*)d_out;

    char* ws = (char*)d_ws;
    f16* wrh = (f16*)(ws + OFF_WR);
    f16* wkh = (f16*)(ws + OFF_WK);
    f16* wvh = (f16*)(ws + OFF_WV);
    f16* woh = (f16*)(ws + OFF_WO);
    f16* xr  = (f16*)(ws + OFF_XR);
    f16* xk  = (f16*)(ws + OFF_XK);
    f16* xv  = (f16*)(ws + OFF_XV);
    float* rbuf = (float*)(ws + OFF_R);
    float* kbuf = (float*)(ws + OFF_K);
    float* vbuf = (float*)(ws + OFF_V);
    float* abuf = (float*)(ws + OFF_A);
    f16*   ybuf = (f16*)(ws + OFF_Y);
    float* zend  = (float*)(ws + OFF_ZE);
    float* carry = (float*)(ws + OFF_CA);
    float* psum  = (float*)(ws + OFF_PS);
    float* stats = (float*)(ws + OFF_ST);

    // 1) weights to f16
    convw_kernel<<<(Dn * Dn) / 256, 256, 0, stream>>>(Wr, Wk, Wv, Wo,
                                                      wrh, wkh, wvh, woh);
    // 2) token-shift mix
    mix_kernel<<<(unsigned)(XE / 256), 256, 0, stream>>>(x, mr, mk, mv, xr, xk, xv);

    // 3) projections: grid (N/256, M/128) = (4, 128), 256 threads
    dim3 ggrid(Dn / 256, Mrows / 128);
    gemm_f16_wT_kernel<<<ggrid, 256, 0, stream>>>(xr, wrh, rbuf, Mrows, Dn, Dn, 1);
    gemm_f16_wT_kernel<<<ggrid, 256, 0, stream>>>(xk, wkh, kbuf, Mrows, Dn, Dn, 0);
    gemm_f16_wT_kernel<<<ggrid, 256, 0, stream>>>(xv, wvh, vbuf, Mrows, Dn, Dn, 0);

    // 4) WKV scan (split-T linear recurrence) + GroupNorm stats
    scan_phase1<<<Bn * Hn * Sn, 64, 0, stream>>>(kbuf, vbuf, td, zend);
    scan_phase2<<<(Bn * Hn * Kn) / 256, 256, 0, stream>>>(td, zend, carry);
    scan_phase3<<<Bn * Hn * Sn, 64, 0, stream>>>(rbuf, kbuf, vbuf, td, tf,
                                                 carry, abuf, psum);
    scan_stats<<<1, 64, 0, stream>>>(psum, stats);

    // 5) GroupNorm apply
    norm_kernel<<<(unsigned)(XE / 256), 256, 0, stream>>>(abuf, stats, gnw, gnb, ybuf);

    // 6) output projection
    gemm_f16_wT_kernel<<<ggrid, 256, 0, stream>>>(ybuf, woh, out, Mrows, Dn, Dn, 0);
}
